// FocusedLinearAttention_420906795696
// MI455X (gfx1250) — compile-verified, matrix operand order checked by reading, not run on hardware
//
#include <hip/hip_runtime.h>
#include <cstdint>
#include <cstddef>

typedef __bf16 bf16;
typedef __attribute__((ext_vector_type(16))) __bf16 v16bf;
typedef __attribute__((ext_vector_type(8)))  float  v8f;

#define DEV static __device__ __forceinline__

DEV uint16_t f2bf_bits(float f) {
  union { float f; uint32_t u; } a; a.f = f;
  uint32_t r = a.u + 0x7FFFu + ((a.u >> 16) & 1u);  // round-to-nearest-even
  return (uint16_t)(r >> 16);
}
DEV bf16 f2bf(float f) {
  union { uint16_t s; bf16 b; } o; o.s = f2bf_bits(f);
  return o.b;
}
DEV float bf2f(bf16 b) {
  union { uint16_t s; bf16 b; } i; i.b = b;
  union { uint32_t u; float f; } o; o.u = ((uint32_t)i.s) << 16;
  return o.f;
}
DEV v16bf load16_bf(const bf16* p) { return *(const v16bf*)p; }

DEV v8f wmma_bf16(v16bf a, v16bf b, v8f c) {
  // D = A(16x32 bf16) x B(32x16 bf16) + C(16x16 f32)
  return __builtin_amdgcn_wmma_f32_16x16x32_bf16(false, a, false, b, (short)0, c,
                                                 false, false);
}

// ---------------------------------------------------------------------------
// Bulk f32 -> bf16 conversion: 8 elements / thread, 128-bit in, 128-bit out.
// ---------------------------------------------------------------------------
__global__ __launch_bounds__(256)
void cvt_bf16_kern(const float* __restrict__ in, uint32_t* __restrict__ out) {
  size_t i = ((size_t)blockIdx.x * 256 + threadIdx.x) * 8;
  const float4 lo = *(const float4*)(in + i);
  const float4 hi = *(const float4*)(in + i + 4);
  uint4 o;
  o.x = (uint32_t)f2bf_bits(lo.x) | ((uint32_t)f2bf_bits(lo.y) << 16);
  o.y = (uint32_t)f2bf_bits(lo.z) | ((uint32_t)f2bf_bits(lo.w) << 16);
  o.z = (uint32_t)f2bf_bits(hi.x) | ((uint32_t)f2bf_bits(hi.y) << 16);
  o.w = (uint32_t)f2bf_bits(hi.z) | ((uint32_t)f2bf_bits(hi.w) << 16);
  *(uint4*)(out + i / 2) = o;
}

// ---------------------------------------------------------------------------
// Pure-bf16 GEMM: Out[m,n] = sum_k X[m,k] * W[n,k]   (X @ W.T)
// block = 256 thr (8 waves); block tile 256(M) x 64(N); wave tile 32 x 64.
// Inner loop: 2 A-loads + 4 B-loads (32B each) -> 8 WMMA.
// ---------------------------------------------------------------------------
__global__ __launch_bounds__(256)
void gemm_bf16_wmma(const bf16* __restrict__ X, const bf16* __restrict__ W,
                    float* __restrict__ Out, int N, int K) {
  const int lane = threadIdx.x & 31;
  const int wave = threadIdx.x >> 5;
  const int m0 = blockIdx.x * 256 + wave * 32;
  const int n0 = blockIdx.y * 64;
  const int mr0 = m0 + (lane & 15);
  const int koff = (lane >> 4) * 16;
  v8f acc[2][4] = {};
  for (int k = 0; k < K; k += 32) {
    v16bf a0 = load16_bf(X + (size_t)mr0 * K + k + koff);
    v16bf a1 = load16_bf(X + (size_t)(mr0 + 16) * K + k + koff);
#pragma unroll
    for (int t = 0; t < 4; ++t) {
      v16bf b = load16_bf(W + (size_t)(n0 + t * 16 + (lane & 15)) * K + k + koff);
      acc[0][t] = wmma_bf16(a0, b, acc[0][t]);
      acc[1][t] = wmma_bf16(a1, b, acc[1][t]);
    }
  }
#pragma unroll
  for (int h = 0; h < 2; ++h) {
    const int rbase = m0 + h * 16 + 8 * (lane >> 4);
#pragma unroll
    for (int t = 0; t < 4; ++t)
#pragma unroll
      for (int r = 0; r < 8; ++r)
        Out[(size_t)(rbase + r) * N + n0 + t * 16 + (lane & 15)] = acc[h][t][r];
  }
}

// Same with bias epilogue (final projection).
__global__ __launch_bounds__(256)
void gemm_bf16_bias_wmma(const bf16* __restrict__ X, const bf16* __restrict__ W,
                         const float* __restrict__ bias, float* __restrict__ Out,
                         int N, int K) {
  const int lane = threadIdx.x & 31;
  const int wave = threadIdx.x >> 5;
  const int m0 = blockIdx.x * 256 + wave * 32;
  const int n0 = blockIdx.y * 64;
  const int mr0 = m0 + (lane & 15);
  const int koff = (lane >> 4) * 16;
  v8f acc[2][4] = {};
  for (int k = 0; k < K; k += 32) {
    v16bf a0 = load16_bf(X + (size_t)mr0 * K + k + koff);
    v16bf a1 = load16_bf(X + (size_t)(mr0 + 16) * K + k + koff);
#pragma unroll
    for (int t = 0; t < 4; ++t) {
      v16bf b = load16_bf(W + (size_t)(n0 + t * 16 + (lane & 15)) * K + k + koff);
      acc[0][t] = wmma_bf16(a0, b, acc[0][t]);
      acc[1][t] = wmma_bf16(a1, b, acc[1][t]);
    }
  }
#pragma unroll
  for (int h = 0; h < 2; ++h) {
    const int rbase = m0 + h * 16 + 8 * (lane >> 4);
#pragma unroll
    for (int t = 0; t < 4; ++t) {
      const int col = n0 + t * 16 + (lane & 15);
      const float bv = bias[col];
#pragma unroll
      for (int r = 0; r < 8; ++r)
        Out[(size_t)(rbase + r) * N + col] = acc[h][t][r] + bv;
    }
  }
}

// ---------------------------------------------------------------------------
// q/k focusing prep: per (b,n) row of kv (1280 f32) -> q,k with softplus scale,
// relu+eps, cube + renorm over C=640. Writes q (head-split bf16) and kT
// (head-split, transposed HDxN bf16).
// ---------------------------------------------------------------------------
__global__ __launch_bounds__(256)
void qk_prep(const float* __restrict__ kvbuf, const float* __restrict__ pos,
             const float* __restrict__ scale,
             bf16* __restrict__ qout, bf16* __restrict__ kTout) {
  const int row = blockIdx.x;          // b*4096 + n
  const int n = row & 4095;
  const int b = row >> 12;
  const int tid = threadIdx.x;
  const float* kvr = kvbuf + (size_t)row * 1280;
  float qv[3], kvv[3];
  float q2 = 0.f, q6 = 0.f, k2 = 0.f, k6 = 0.f;
#pragma unroll
  for (int i = 0; i < 3; ++i) {
    int c = tid + i * 256;
    float qq = 0.f, kk = 0.f;
    if (c < 640) {
      float sc = log1pf(expf(scale[c]));               // softplus
      float kraw = kvr[c] + pos[(size_t)n * 640 + c];  // k = kv[...,0,:] + pos
      float qraw = kvr[640 + c];                       // q = kv[...,1,:]
      qq = (fmaxf(qraw, 0.f) + 1e-6f) / sc;
      kk = (fmaxf(kraw, 0.f) + 1e-6f) / sc;
    }
    qv[i] = qq; kvv[i] = kk;
    q2 += qq * qq; k2 += kk * kk;
    float q3 = qq * qq * qq, k3 = kk * kk * kk;
    q6 += q3 * q3; k6 += k3 * k3;
  }
  __shared__ float r0[256], r1[256], r2[256], r3[256];
  r0[tid] = q2; r1[tid] = q6; r2[tid] = k2; r3[tid] = k6;
  __syncthreads();
  for (int s = 128; s > 0; s >>= 1) {
    if (tid < s) {
      r0[tid] += r0[tid + s]; r1[tid] += r1[tid + s];
      r2[tid] += r2[tid + s]; r3[tid] += r3[tid + s];
    }
    __syncthreads();
  }
  const float qf = sqrtf(r0[0]) * rsqrtf(r1[0] + 1e-30f);  // |q| / |q^3|
  const float kf = sqrtf(r2[0]) * rsqrtf(r3[0] + 1e-30f);
#pragma unroll
  for (int i = 0; i < 3; ++i) {
    int c = tid + i * 256;
    if (c < 640) {
      int h = c >> 7, hd = c & 127;
      int bh = b * 5 + h;
      float q3 = qv[i] * qv[i] * qv[i] * qf;
      float k3 = kvv[i] * kvv[i] * kvv[i] * kf;
      qout[((size_t)bh * 4096 + n) * 128 + hd] = f2bf(q3);
      kTout[((size_t)bh * 128 + hd) * 4096 + n] = f2bf(k3);
    }
  }
}

// v split-heads (bf16) + transposed copy (HDxN per head)
__global__ __launch_bounds__(256)
void v_prep(const float* __restrict__ vin, bf16* __restrict__ vh,
            bf16* __restrict__ vT) {
  size_t idx = (size_t)blockIdx.x * 256 + threadIdx.x;  // over 32768*640
  int c = (int)(idx % 640);
  size_t row = idx / 640;
  int n = (int)(row & 4095), b = (int)(row >> 12);
  int h = c >> 7, hd = c & 127;
  int bh = b * 5 + h;
  bf16 val = f2bf(vin[idx]);
  vh[((size_t)bh * 4096 + n) * 128 + hd] = val;
  vT[((size_t)bh * 128 + hd) * 4096 + n] = val;
}

// ksum[bh,c] = sum_n k[bh,n,c]  (reads contiguous kT rows)
__global__ __launch_bounds__(256)
void ksum_kern(const bf16* __restrict__ kT, float* __restrict__ ksum) {
  const int rc = blockIdx.x;  // bh*128 + c
  const bf16* row = kT + (size_t)rc * 4096;
  float s = 0.f;
  for (int j = threadIdx.x; j < 4096; j += 256) s += bf2f(row[j]);
  __shared__ float red[256];
  red[threadIdx.x] = s; __syncthreads();
  for (int st = 128; st > 0; st >>= 1) {
    if (threadIdx.x < st) red[threadIdx.x] += red[threadIdx.x + st];
    __syncthreads();
  }
  if (threadIdx.x == 0) ksum[rc] = red[0];
}

// z[bh,n] = 1 / (q[bh,n,:] . ksum[bh,:] + eps)
__global__ __launch_bounds__(256)
void z_kern(const bf16* __restrict__ q, const float* __restrict__ ksum,
            float* __restrict__ z) {
  size_t idx = (size_t)blockIdx.x * 256 + threadIdx.x;  // bh*4096 + n
  int bh = (int)(idx >> 12);
  const bf16* qr = q + idx * 128;
  const float* ks = ksum + (size_t)bh * 128;
  float d = 0.f;
#pragma unroll 8
  for (int c = 0; c < 128; ++c) d += bf2f(qr[c]) * ks[c];
  z[idx] = 1.f / (d + 1e-6f);
}

// kv2T[bh][d][c] = sum_j vT[bh][d][j] * kT[bh][c][j]   (128x128, K=4096)
// 1 block per head; wave w -> rows d in [16w,16w+16), 8 col-tiles, 8 accums.
__global__ __launch_bounds__(256)
void kv2_kern(const bf16* __restrict__ vT, const bf16* __restrict__ kT,
              bf16* __restrict__ kv2T) {
  const int bh = blockIdx.x;
  const int lane = threadIdx.x & 31, wave = threadIdx.x >> 5;
  const int d0 = wave * 16;
  const int koff = (lane >> 4) * 16;
  const bf16* vbase = vT + (size_t)bh * 128 * 4096;
  const bf16* kbase = kT + (size_t)bh * 128 * 4096;
  v8f acc[8] = {};
  for (int j = 0; j < 4096; j += 32) {
    v16bf a = load16_bf(vbase + (size_t)(d0 + (lane & 15)) * 4096 + j + koff);
#pragma unroll
    for (int t = 0; t < 8; ++t) {
      v16bf b = load16_bf(kbase + (size_t)(t * 16 + (lane & 15)) * 4096 + j + koff);
      acc[t] = wmma_bf16(a, b, acc[t]);
    }
  }
  bf16* obase = kv2T + (size_t)bh * 128 * 128;
  const int rbase = d0 + 8 * (lane >> 4);
#pragma unroll
  for (int t = 0; t < 8; ++t)
#pragma unroll
    for (int r = 0; r < 8; ++r)
      obase[(size_t)(rbase + r) * 128 + t * 16 + (lane & 15)] = f2bf(acc[t][r]);
}

// x[bh][i][d] = z[bh][i] * sum_c q[bh][i][c] * kv2T[bh][d][c]
__global__ __launch_bounds__(256)
void attn_kern(const bf16* __restrict__ q, const bf16* __restrict__ kv2T,
               const float* __restrict__ z, float* __restrict__ xout) {
  const int bh = blockIdx.x;
  const int lane = threadIdx.x & 31, wave = threadIdx.x >> 5;
  const int m0 = blockIdx.y * 128 + wave * 16;
  const int koff = (lane >> 4) * 16;
  const bf16* qbase = q + (size_t)bh * 4096 * 128;
  const bf16* kvb = kv2T + (size_t)bh * 128 * 128;
  v8f acc[8] = {};
#pragma unroll
  for (int k = 0; k < 128; k += 32) {
    v16bf a = load16_bf(qbase + (size_t)(m0 + (lane & 15)) * 128 + k + koff);
#pragma unroll
    for (int t = 0; t < 8; ++t) {
      v16bf b = load16_bf(kvb + (size_t)(t * 16 + (lane & 15)) * 128 + k + koff);
      acc[t] = wmma_bf16(a, b, acc[t]);
    }
  }
  const int rbase = m0 + 8 * (lane >> 4);
  const float* zb = z + (size_t)bh * 4096;
  float* ob = xout + (size_t)bh * 4096 * 128;
#pragma unroll
  for (int r = 0; r < 8; ++r) {
    const float zr = zb[rbase + r];
#pragma unroll
    for (int t = 0; t < 8; ++t)
      ob[(size_t)(rbase + r) * 128 + t * 16 + (lane & 15)] = acc[t][r] * zr;
  }
}

// 5x5 depthwise conv on v (per head channel), add bias and attention output,
// write merged-head bf16 buffer for the projection GEMM.
__global__ __launch_bounds__(256)
void dwconv_kern(const bf16* __restrict__ vh, const float* __restrict__ wgt,
                 const float* __restrict__ bias, const float* __restrict__ xin,
                 bf16* __restrict__ xpre) {
  size_t idx = (size_t)blockIdx.x * 256 + threadIdx.x;  // bh*4096*128
  int c = (int)(idx & 127);
  int pix = (int)((idx >> 7) & 4095);
  int bh = (int)(idx >> 19);
  int y = pix >> 6, x = pix & 63;
  float acc = bias[c];
  const bf16* vb = vh + ((size_t)bh << 19);
  const float* w = wgt + c * 25;
#pragma unroll
  for (int dy = 0; dy < 5; ++dy) {
    int yy = y + dy - 2;
    if (yy < 0 || yy > 63) continue;
#pragma unroll
    for (int dx = 0; dx < 5; ++dx) {
      int xx = x + dx - 2;
      if (xx < 0 || xx > 63) continue;
      acc += bf2f(vb[(size_t)(yy * 64 + xx) * 128 + c]) * w[dy * 5 + dx];
    }
  }
  float val = xin[idx] + acc;
  int b = bh / 5, h = bh % 5;
  xpre[((size_t)(b * 4096 + pix)) * 640 + h * 128 + c] = f2bf(val);
}

// ---------------------------------------------------------------------------
extern "C" void kernel_launch(void* const* d_in, const int* in_sizes, int n_in,
                              void* d_out, int out_size, void* d_ws, size_t ws_size,
                              hipStream_t stream) {
  (void)in_sizes; (void)n_in; (void)out_size; (void)ws_size;
  const float* x1     = (const float*)d_in[0];
  const float* x2     = (const float*)d_in[1];
  const float* Wq     = (const float*)d_in[2];
  const float* Wkv    = (const float*)d_in[3];
  const float* Wproj  = (const float*)d_in[4];
  const float* bproj  = (const float*)d_in[5];
  const float* dwc_w  = (const float*)d_in[6];
  const float* dwc_b  = (const float*)d_in[7];
  const float* scale  = (const float*)d_in[8];
  const float* pos    = (const float*)d_in[9];
  float* out = (float*)d_out;

  char* ws = (char*)d_ws;
  const size_t NB = 32768;  // B*N
  size_t off = 0;
  auto take = [&](size_t bytes) {
    size_t o = off; off += (bytes + 255) & ~(size_t)255; return o;
  };
  size_t kv_off   = take(NB * 1280 * 4);        // kv = x1 @ Wkv.T (f32)
  size_t v_off    = take(NB * 640 * 4);         // v  = x2 @ Wq.T  (f32)
  size_t x1b_off  = take(NB * 640 * 2);         // x1 bf16
  size_t x2b_off  = take(NB * 640 * 2);         // x2 bf16
  size_t wqb_off  = take((size_t)640 * 640 * 2);
  size_t wkvb_off = take((size_t)1280 * 640 * 2);
  size_t wpb_off  = take((size_t)640 * 640 * 2);
  size_t q_off    = take(NB * 640 * 2);         // q  bf16, head-split (bh,n,hd)
  size_t kT_off   = take(NB * 640 * 2);         // kT bf16, head-split transposed
  size_t vh_off   = take(NB * 640 * 2);         // v  bf16, head-split
  size_t vT_off   = take(NB * 640 * 2);         // vT bf16, head-split transposed
  size_t ksum_off = take(40 * 128 * 4);
  size_t z_off    = take((size_t)40 * 4096 * 4);
  size_t kv2_off  = take((size_t)40 * 128 * 128 * 2);
  size_t x_off    = kv_off;                     // reuse kv region post-prep
  size_t xpre_off = kv_off + NB * 640 * 4;      // still inside old kv region

  float* kvbuf = (float*)(ws + kv_off);
  float* vbuf  = (float*)(ws + v_off);
  bf16*  x1b   = (bf16*)(ws + x1b_off);
  bf16*  x2b   = (bf16*)(ws + x2b_off);
  bf16*  wqb   = (bf16*)(ws + wqb_off);
  bf16*  wkvb  = (bf16*)(ws + wkvb_off);
  bf16*  wpb   = (bf16*)(ws + wpb_off);
  bf16*  qb    = (bf16*)(ws + q_off);
  bf16*  kTb   = (bf16*)(ws + kT_off);
  bf16*  vhb   = (bf16*)(ws + vh_off);
  bf16*  vTb   = (bf16*)(ws + vT_off);
  float* ksumb = (float*)(ws + ksum_off);
  float* zb    = (float*)(ws + z_off);
  bf16*  kv2b  = (bf16*)(ws + kv2_off);
  float* xb    = (float*)(ws + x_off);
  bf16*  xpre  = (bf16*)(ws + xpre_off);

  dim3 blk(256);
  // f32 -> bf16 staging (8 elems / thread)
  cvt_bf16_kern<<<10240, blk, 0, stream>>>(x1, (uint32_t*)x1b);
  cvt_bf16_kern<<<10240, blk, 0, stream>>>(x2, (uint32_t*)x2b);
  cvt_bf16_kern<<<200,   blk, 0, stream>>>(Wq, (uint32_t*)wqb);
  cvt_bf16_kern<<<400,   blk, 0, stream>>>(Wkv, (uint32_t*)wkvb);
  cvt_bf16_kern<<<200,   blk, 0, stream>>>(Wproj, (uint32_t*)wpb);
  // GEMMs (pure bf16 inner loops)
  gemm_bf16_wmma<<<dim3(128, 10), blk, 0, stream>>>(x2b, wqb, vbuf, 640, 640);
  gemm_bf16_wmma<<<dim3(128, 20), blk, 0, stream>>>(x1b, wkvb, kvbuf, 1280, 640);
  // focusing nonlinearity + layout prep
  qk_prep<<<32768, blk, 0, stream>>>(kvbuf, pos, scale, qb, kTb);
  v_prep<<<81920, blk, 0, stream>>>(vbuf, vhb, vTb);
  ksum_kern<<<5120, blk, 0, stream>>>(kTb, ksumb);
  z_kern<<<640, blk, 0, stream>>>(qb, ksumb, zb);
  // linear attention
  kv2_kern<<<40, blk, 0, stream>>>(vTb, kTb, kv2b);
  attn_kern<<<dim3(40, 32), blk, 0, stream>>>(qb, kv2b, zb, xb);
  // depthwise conv + merge heads
  dwconv_kern<<<81920, blk, 0, stream>>>(vhb, dwc_w, dwc_b, xb, xpre);
  // output projection
  gemm_bf16_bias_wmma<<<dim3(128, 10), blk, 0, stream>>>(xpre, wpb, bproj, out,
                                                         640, 640);
}
